// VGAEEncoder_71914932404372
// MI455X (gfx1250) — compile-verified, matrix operand order checked by reading, not run on hardware
//
#include <hip/hip_runtime.h>
#include <hip/hip_bf16.h>

// ---------------------------------------------------------------------------
// VGAE encoder (NNConv) fused pipeline for gfx1250 (MI455X).
// All GEMMs use v_wmma_f32_16x16x32_bf16 (bf16 in, f32 accumulate).
// The per-edge weight matrices W_e (1.64 GB if materialized) are generated in
// registers and contracted immediately -> never written to HBM.
// h[src] gather uses global_load_async_to_lds (ASYNCcnt) into a transposed
// LDS tile; WMMA operands come from pre-packed bf16-pair dwords.
// ---------------------------------------------------------------------------

typedef __attribute__((ext_vector_type(16))) __bf16 v16bf;
typedef __attribute__((ext_vector_type(8)))  float  v8f;

union BF16x16 { v16bf v; unsigned short s[16]; unsigned int u[8]; };

#if defined(__AMDGCN__) && __has_builtin(__builtin_amdgcn_global_load_async_to_lds_b32) && __has_builtin(__builtin_amdgcn_s_wait_asynccnt)
#define HAVE_ASYNC_LDS 1
#else
#define HAVE_ASYNC_LDS 0
#endif

// float -> bf16 round-to-nearest-even via bit manipulation (fallback)
__device__ __forceinline__ unsigned short f2bf(float f) {
    union { float f; unsigned int u; } c; c.f = f;
    unsigned int u = c.u;
    u += 0x7FFFu + ((u >> 16) & 1u);
    return (unsigned short)(u >> 16);
}

// pack two floats into a bf16-pair dword (low = first arg)
__device__ __forceinline__ unsigned int pack2(float a, float b) {
#if defined(__AMDGCN__) && __has_builtin(__builtin_amdgcn_cvt_pk_bf16_f32)
    typedef __attribute__((ext_vector_type(2))) __bf16 v2bf;
    union { v2bf v; unsigned int u; } r;
    r.v = __builtin_amdgcn_cvt_pk_bf16_f32(a, b);
    return r.u;
#else
    return (unsigned int)f2bf(a) | ((unsigned int)f2bf(b) << 16);
#endif
}

// K index held by VGPR-pair i (0..7) for a lane in half `hi` (0/1), per the
// CDNA5 16-bit A/B 16x32 register layout (ISA 7.12.2)
__device__ __forceinline__ int kmap(int i, int hi) {
    return (i < 4 ? 2 * i : 16 + 2 * (i - 4)) + 8 * hi;
}

__device__ __forceinline__ v8f wmma_bf16(const BF16x16& a, const BF16x16& b, v8f c) {
    return __builtin_amdgcn_wmma_f32_16x16x32_bf16(
        /*neg_a=*/false, a.v, /*neg_b=*/false, b.v,
        /*c_mod=*/(short)0, c, /*reuse_a=*/false, /*reuse_b=*/false);
}

// ---------------------------------------------------------------------------
// Prep kernels: run every launch (deterministic, no cached state).
// ---------------------------------------------------------------------------
__global__ void zero_kernel(float* __restrict__ p, int n) {
    int i = blockIdx.x * blockDim.x + threadIdx.x;
    if (i < n) p[i] = 0.0f;
}

// adjacent-pair pack along the contiguous (last) dim
__global__ void pack_pairs_kernel(const float* __restrict__ src,
                                  unsigned int* __restrict__ dst, int npairs) {
    int i = blockIdx.x * blockDim.x + threadIdx.x;
    if (i < npairs) dst[i] = pack2(src[2 * i], src[2 * i + 1]);
}

// weight transpose-pack: W[K][C] row-major -> wp[C][K/2] with pairs along K
__global__ void pack_wT_kernel(const float* __restrict__ W,
                               unsigned int* __restrict__ wp, int K, int C) {
    int i = blockIdx.x * blockDim.x + threadIdx.x;
    int K2 = K >> 1;
    if (i >= C * K2) return;
    int c = i / K2, j = i % K2;
    wp[i] = pack2(W[(2 * j) * C + c], W[(2 * j + 1) * C + c]);
}

// bf16 in low 16 bits (used as the bias K-row of the B matrix)
__global__ void pack_lo_kernel(const float* __restrict__ src,
                               unsigned int* __restrict__ dst, int n) {
    int i = blockIdx.x * blockDim.x + threadIdx.x;
    if (i < n) dst[i] = pack2(src[i], 0.0f) & 0xFFFFu;
}

// ---------------------------------------------------------------------------
// Kernel 1: h = relu(x @ lin_in_w + lin_in_b)   [20000,128] x [128,64]
// One wave computes one 16x16 output tile; K=128 -> 4 WMMA steps.
// ---------------------------------------------------------------------------
__global__ void lin_in_kernel(const float* __restrict__ x,
                              const unsigned int* __restrict__ wpl, // [64][64] pairs
                              const float* __restrict__ bias,
                              float* __restrict__ h, int N) {
    const int lane = threadIdx.x & 31;
    const int wave = threadIdx.x >> 5;
    const int tile = blockIdx.x * (blockDim.x >> 5) + wave;
    const int NT   = 4;
    const int mt   = tile / NT;
    const int nt   = tile % NT;
    if (mt * 16 >= N) return;         // uniform per wave

    const int m   = lane & 15;
    const int hi  = lane >> 4;
    const int row = mt * 16 + m;
    const int col = nt * 16 + m;

    v8f acc = {};
    for (int kk = 0; kk < 4; ++kk) {
        BF16x16 A, B;
#pragma unroll
        for (int i = 0; i < 8; ++i) {
            int k = kk * 32 + kmap(i, hi);
            A.u[i] = pack2(x[row * 128 + k], x[row * 128 + k + 1]);
            B.u[i] = wpl[col * 64 + (k >> 1)];
        }
        acc = wmma_bf16(A, B, acc);
    }
#pragma unroll
    for (int v = 0; v < 8; ++v) {
        int mo = v + 8 * hi;
        float r = acc[v] + bias[col];
        h[(mt * 16 + mo) * 64 + col] = r > 0.0f ? r : 0.0f;
    }
}

// ---------------------------------------------------------------------------
// Kernel 2 (fused core): per 16-edge group, generate relu(edge MLP) tiles via
// WMMA and immediately contract against h[src]; atomically scatter into the
// segment-sum buffers. Bias folded into WMMA via ones-column at K=16.
// One wave (32 threads) per group.
// ---------------------------------------------------------------------------
#define HT_STRIDE 20   // padded row stride (dwords): 16B-aligned rows, low conflicts

__global__ void edge_msg_kernel(const float* __restrict__ h,
                                const int*   __restrict__ ei,   // [2*E] src|dst
                                const unsigned int* __restrict__ eap, // [E][8] pairs
                                const unsigned int* __restrict__ kwp, // [4096][8] pairs
                                const unsigned int* __restrict__ kbp, // [4096] bf16-lo
                                float* __restrict__ num,        // [N,64] accum
                                float* __restrict__ cnt,        // [N]    accum
                                int E) {
    __shared__ float hsrcT[64 * HT_STRIDE]; // hsrcT[k][e] = h[src[e]][k]
    __shared__ int   ssrc[16];
    __shared__ int   sdst[16];

    const int lane = threadIdx.x;     // 32 threads = 1 wave
    const int e0   = blockIdx.x * 16;
    const int m    = lane & 15;
    const int hi   = lane >> 4;

    // prefetch next group's packed edge attributes (global_prefetch path)
    if (e0 + 16 < E) __builtin_prefetch(&eap[(e0 + 16) * 8], 0, 1);

    if (lane < 16) {
        int e = e0 + lane;
        if (e < E) {
            ssrc[lane] = ei[e];
            int d = ei[E + e];
            sdst[lane] = d;
            atomicAdd(&cnt[d], 1.0f);      // scatter-mean denominator
        } else {
            ssrc[lane] = 0;
            sdst[lane] = 0;
        }
    }
    __syncthreads();

    // ---- gather h[src[e]] rows directly into transposed LDS tile ----------
    // per j: lane writes hsrcT[k= 2j+hi][e=m] = h[src[e]*64 + k]
    {
        const int srcn = ssrc[m];
        const bool ok  = (e0 + m) < E;
#if HAVE_ASYNC_LDS
        int* hnc = (int*)const_cast<float*>(h);
#pragma unroll 4
        for (int j = 0; j < 32; ++j) {
            int k = 2 * j + hi;
            if (ok) {
                __builtin_amdgcn_global_load_async_to_lds_b32(
                    (__attribute__((address_space(1))) int*)(hnc + (size_t)srcn * 64 + k),
                    (__attribute__((address_space(3))) int*)&hsrcT[k * HT_STRIDE + m],
                    0, 0);
            } else {
                hsrcT[k * HT_STRIDE + m] = 0.0f;
            }
        }
        __builtin_amdgcn_s_wait_asynccnt(0);
#else
#pragma unroll 4
        for (int j = 0; j < 32; ++j) {
            int k = 2 * j + hi;
            hsrcT[k * HT_STRIDE + m] = ok ? h[(size_t)srcn * 64 + k] : 0.0f;
        }
#endif
    }
    __syncthreads();

    // ---- A tile: edge_attr rows, K=16 real + ones-column at K=16 ----------
    BF16x16 A;
    {
        uint4 at = {0u, 0u, 0u, 0u};
        if (e0 + m < E) at = ((const uint4*)eap)[(e0 + m) * 2 + hi];
        A.u[0] = at.x; A.u[1] = at.y; A.u[2] = at.z; A.u[3] = at.w;
        A.u[4] = (hi == 0) ? 0x00003F80u : 0u;    // K=16: bf16(1.0)
        A.u[5] = 0u; A.u[6] = 0u; A.u[7] = 0u;
    }

    float macc[4][8];
#pragma unroll
    for (int ob = 0; ob < 4; ++ob)
#pragma unroll
        for (int v = 0; v < 8; ++v) macc[ob][v] = 0.0f;

    for (int hin = 0; hin < 64; ++hin) {
        // 8 scale factors for this lane's 8 edges: two aligned ds_load_b128
        const float4 ha = *(const float4*)&hsrcT[hin * HT_STRIDE + 8 * hi];
        const float4 hb = *(const float4*)&hsrcT[hin * HT_STRIDE + 8 * hi + 4];
        const float hs[8] = {ha.x, ha.y, ha.z, ha.w, hb.x, hb.y, hb.z, hb.w};

        for (int ob = 0; ob < 4; ++ob) {
            const int col = hin * 64 + ob * 16 + m;   // column of flattened 64x64 W_e
            BF16x16 B;
            uint4 bt = ((const uint4*)kwp)[col * 2 + hi];
            B.u[0] = bt.x; B.u[1] = bt.y; B.u[2] = bt.z; B.u[3] = bt.w;
            B.u[4] = (hi == 0) ? kbp[col] : 0u;       // K=16: bias row
            B.u[5] = 0u; B.u[6] = 0u; B.u[7] = 0u;
            v8f c = {};
            v8f d = wmma_bf16(A, B, c);               // = EA*W + bias
#pragma unroll
            for (int v = 0; v < 8; ++v) {
                float w = d[v];
                w = w > 0.0f ? w : 0.0f;              // relu on the edge kernel
                macc[ob][v] += w * hs[v];             // contract with h[src]
            }
        }
    }

    // scatter-add messages into destination nodes
#pragma unroll
    for (int ob = 0; ob < 4; ++ob)
#pragma unroll
        for (int v = 0; v < 8; ++v) {
            int e = v + 8 * hi;
            if (e0 + e < E)
                atomicAdd(&num[(size_t)sdst[e] * 64 + ob * 16 + m], macc[ob][v]);
        }
}

// ---------------------------------------------------------------------------
// Kernel 3: h2 = relu(h @ root_w + num/max(cnt,1) + conv_b), then
//           mu = h2 @ mu_w + mu_b ; lv = h2 @ lv_w + lv_b.
// One wave per 16-node tile; h2 lives only in LDS.
// ---------------------------------------------------------------------------
__global__ void finalize_kernel(const float* __restrict__ h,
                                const float* __restrict__ num,
                                const float* __restrict__ cnt,
                                const unsigned int* __restrict__ rwp,  // [64][32] pairs
                                const float* __restrict__ cb,
                                const unsigned int* __restrict__ muwp, // [32][32] pairs
                                const float* __restrict__ mub,
                                const unsigned int* __restrict__ lvwp, // [32][32] pairs
                                const float* __restrict__ lvb,
                                float* __restrict__ out, int N) {
    __shared__ float h2[16][64];

    const int lane = threadIdx.x;
    const int mt   = blockIdx.x;
    const int m    = lane & 15;
    const int hi   = lane >> 4;
    const int row  = mt * 16 + m;

    // root transform: [16,64] x [64,64], K=64 -> 2 WMMA steps per column tile
    for (int nt = 0; nt < 4; ++nt) {
        const int col = nt * 16 + m;
        v8f acc = {};
        for (int kk = 0; kk < 2; ++kk) {
            BF16x16 A, B;
#pragma unroll
            for (int i = 0; i < 8; ++i) {
                int k = kk * 32 + kmap(i, hi);
                A.u[i] = pack2(h[row * 64 + k], h[row * 64 + k + 1]);
                B.u[i] = rwp[col * 32 + (k >> 1)];
            }
            acc = wmma_bf16(A, B, acc);
        }
#pragma unroll
        for (int v = 0; v < 8; ++v) {
            int mo   = v + 8 * hi;
            int node = mt * 16 + mo;
            float c  = cnt[node];
            c = c > 1.0f ? c : 1.0f;
            float r = acc[v] + num[(size_t)node * 64 + col] / c + cb[col];
            h2[mo][col] = r > 0.0f ? r : 0.0f;
        }
    }
    __syncthreads();

    // output heads: [16,64] x [64,32] each, K=64 -> 2 WMMA steps, 2 col tiles
    for (int head = 0; head < 2; ++head) {
        const unsigned int* Wh = head ? lvwp : muwp;
        const float* bh = head ? lvb : mub;
        float* oh = out + (size_t)head * N * 32;
        for (int nt = 0; nt < 2; ++nt) {
            const int col = nt * 16 + m;
            v8f acc = {};
            for (int kk = 0; kk < 2; ++kk) {
                BF16x16 A, B;
#pragma unroll
                for (int i = 0; i < 8; ++i) {
                    int k = kk * 32 + kmap(i, hi);
                    A.u[i] = pack2(h2[m][k], h2[m][k + 1]);
                    B.u[i] = Wh[col * 32 + (k >> 1)];
                }
                acc = wmma_bf16(A, B, acc);
            }
#pragma unroll
            for (int v = 0; v < 8; ++v) {
                int mo = v + 8 * hi;
                oh[(size_t)(mt * 16 + mo) * 32 + col] = acc[v] + bh[col];
            }
        }
    }
}

// ---------------------------------------------------------------------------
extern "C" void kernel_launch(void* const* d_in, const int* in_sizes, int n_in,
                              void* d_out, int out_size, void* d_ws, size_t ws_size,
                              hipStream_t stream) {
    const float* x   = (const float*)d_in[0];
    const int*   ei  = (const int*)  d_in[1];
    const float* ea  = (const float*)d_in[2];
    const float* liw = (const float*)d_in[3];
    const float* lib = (const float*)d_in[4];
    const float* kw  = (const float*)d_in[5];
    const float* kb  = (const float*)d_in[6];
    const float* rw  = (const float*)d_in[7];
    const float* cb  = (const float*)d_in[8];
    const float* muw = (const float*)d_in[9];
    const float* mub = (const float*)d_in[10];
    const float* lvw = (const float*)d_in[11];
    const float* lvb = (const float*)d_in[12];
    float* out = (float*)d_out;

    const int N = in_sizes[0] / 128;   // 20000
    const int E = in_sizes[2] / 16;    // 100000

    // workspace layout
    char* wsp = (char*)d_ws;
    float* h_buf = (float*)wsp;                 wsp += (size_t)N * 64 * 4;
    float* numb  = (float*)wsp;                 wsp += (size_t)N * 64 * 4;
    float* cntb  = (float*)wsp;                 wsp += (size_t)N * 4;
    unsigned int* eap  = (unsigned int*)wsp;    wsp += (size_t)E * 8 * 4;
    unsigned int* kwp  = (unsigned int*)wsp;    wsp += (size_t)4096 * 8 * 4;
    unsigned int* kbp  = (unsigned int*)wsp;    wsp += (size_t)4096 * 4;
    unsigned int* wpl  = (unsigned int*)wsp;    wsp += (size_t)64 * 64 * 4;
    unsigned int* rwp  = (unsigned int*)wsp;    wsp += (size_t)64 * 32 * 4;
    unsigned int* muwp = (unsigned int*)wsp;    wsp += (size_t)32 * 32 * 4;
    unsigned int* lvwp = (unsigned int*)wsp;    wsp += (size_t)32 * 32 * 4;

    // --- prep: pack bf16 operands (recomputed every launch, deterministic) ---
    const int eaP = E * 8;
    pack_pairs_kernel<<<(eaP + 255) / 256, 256, 0, stream>>>(ea, eap, eaP);
    pack_wT_kernel<<<(4096 * 8 + 255) / 256, 256, 0, stream>>>(kw, kwp, 16, 4096);
    pack_lo_kernel<<<(4096 + 255) / 256, 256, 0, stream>>>(kb, kbp, 4096);
    pack_wT_kernel<<<(64 * 64 + 255) / 256, 256, 0, stream>>>(liw, wpl, 128, 64);
    pack_wT_kernel<<<(64 * 32 + 255) / 256, 256, 0, stream>>>(rw, rwp, 64, 64);
    pack_wT_kernel<<<(32 * 32 + 255) / 256, 256, 0, stream>>>(muw, muwp, 64, 32);
    pack_wT_kernel<<<(32 * 32 + 255) / 256, 256, 0, stream>>>(lvw, lvwp, 64, 32);

    // zero the atomic accumulators every launch (num and cnt are contiguous)
    const int zn = N * 64 + N;
    zero_kernel<<<(zn + 255) / 256, 256, 0, stream>>>(numb, zn);

    // h = relu(x @ lin_in_w + b): tiles = (N/16) * 4, 4 waves per block
    const int mtiles = (N + 15) / 16;
    const int tiles  = mtiles * 4;
    lin_in_kernel<<<(tiles + 3) / 4, 128, 0, stream>>>(x, wpl, lib, h_buf, N);

    // fused per-edge kernel-MLP + message + scatter
    const int groups = (E + 15) / 16;
    edge_msg_kernel<<<groups, 32, 0, stream>>>(h_buf, ei, eap, kwp, kbp, numb, cntb, E);

    // root transform + mean aggregate + output heads
    finalize_kernel<<<mtiles, 32, 0, stream>>>(h_buf, numb, cntb, rwp, cb,
                                               muwp, mub, lvwp, lvb, out, N);
}
